// NATBlock_43954695307303
// MI455X (gfx1250) — compile-verified, matrix operand order checked by reading, not run on hardware
//
#include <hip/hip_runtime.h>
#include <math.h>

// ---------------- problem constants ----------------
constexpr int BB   = 2;
constexpr int LL   = 512;
constexpr int DD   = 128;
constexpr int HH   = 4;
constexpr int HD   = 32;      // head dim
constexpr int HID  = 512;     // mlp hidden
constexpr long NX  = (long)BB * LL * DD;          // 131072 (one x tensor)
constexpr long BIAS1 = (long)BB * HH * LL * LL;   // one layer's bias tensor

typedef __attribute__((ext_vector_type(16))) _Float16 v16h;
typedef __attribute__((ext_vector_type(4)))  _Float16 v4h;
typedef __attribute__((ext_vector_type(8)))  float    v8f;

#define DEV __device__ __forceinline__

// K index of element j in a 16-bit A/B WMMA fragment (16x32 / 32x16), wave32.
DEV int frag_k(int j, int half) { return (j & 7) + ((j >> 3) << 4) + half * 8; }

DEV v8f wmma_f16(v16h a, v16h b, v8f c) {
    return __builtin_amdgcn_wmma_f32_16x16x32_f16(false, a, false, b, (short)0, c,
                                                  false, false);
}

DEV v4h cvt4(float4 f) {
    v4h h;
    h[0] = (_Float16)f.x; h[1] = (_Float16)f.y;
    h[2] = (_Float16)f.z; h[3] = (_Float16)f.w;
    return h;
}

// =====================================================================
// 0) weight pre-swizzle: fp32 weights -> fragment-major f16 so each WMMA
//    B fragment is ONE contiguous 32-byte v16h load.
//    dst[((kt*NT + nt)*32 + lane)*16 + j] = W[kt*32 + frag_k(j,lane>>4)][nt*16 + lane&15]
// =====================================================================
__global__ __launch_bounds__(256) void prep_w_kernel(
    int mode, const float* __restrict__ s0, const float* __restrict__ s1,
    const float* __restrict__ s2, _Float16* __restrict__ dst, int K, int N)
{
    int s = blockIdx.x * 256 + threadIdx.x;
    if (s >= K * N) return;
    int j = s & 15, lane = (s >> 4) & 31, tile = s >> 9;
    int ntiles = N >> 4;
    int nt = tile % ntiles, kt = tile / ntiles;
    int row = kt * 32 + frag_k(j, lane >> 4);
    int col = nt * 16 + (lane & 15);
    float v = 0.f;
    switch (mode) {
    case 0:   // plain W[K][N]
        v = s0[(long)row * N + col]; break;
    case 1:   // fused qkv: cols 0-127 wq, 128-255 wk, 256-383 wv (each [128][128])
        v = (col < 128) ? s0[(long)row * 128 + col]
          : (col < 256) ? s1[(long)row * 128 + col - 128]
                        : s2[(long)row * 128 + col - 256];
        break;
    case 2:   // rpe pack: cols 0-3 layer0 w_rpe[128][4], 4-7 layer1, 8-15 zero
        if (col < 4)      v = s0[row * 4 + col];
        else if (col < 8) v = s1[row * 4 + col - 4];
        break;
    case 3: { // conv1d [co][ci][kw], K index = kw*128+ci
        int kw = row >> 7, ci = row & 127;
        v = s0[((long)col * 128 + ci) * 3 + kw];
    } break;
    case 4: { // conv2d [co][ci][kh][kw], K index = (kh*3+kw)*128+ci
        int kh = row / 384, rem = row % 384, kw = rem >> 7, ci = rem & 127;
        v = s0[(((long)col * 128 + ci) * 3 + kh) * 3 + kw];
    } break;
    }
    dst[s] = (_Float16)v;
}

// =====================================================================
// 1) bias projection: bias[layer,b,h,l,m] = rpe[b,l,m,:] . w_rpe[:,h] + b_rpe
//    ONE pass over the 268MB rpe tensor for BOTH layers. Tile rows staged
//    through wave-private LDS with coalesced b128 loads; 4 B fragments live
//    in VGPRs for the whole kernel.
// =====================================================================
__global__ __launch_bounds__(128) void bias_proj_kernel(
    const float* __restrict__ rpe, const _Float16* __restrict__ Wr,
    const float* __restrict__ b0, const float* __restrict__ b1,
    float* __restrict__ bias_all)
{
    __shared__ float stg[4][16][DD];   // 32 KB, wave-private slices
    int tid = threadIdx.x, wave = tid >> 5, lane = tid & 31;
    int lm = lane & 15, lh = lane >> 4;

    v16h bf[4];
    for (int ks = 0; ks < 4; ++ks)
        bf[ks] = *(const v16h*)(Wr + (long)ks * 512 + lane * 16);

    for (int t = 0; t < 4; ++t) {
        long tile = ((long)blockIdx.x * 4 + wave) * 4 + t;  // 32768 tiles
        long rb   = tile * 16;
        long bl   = rb >> 9;
        int  b    = (int)(bl >> 9);
        int  l    = (int)(bl & 511);
        int  m0   = (int)(rb & 511);

        // stage 16x128 fp32 rows, coalesced float4 per lane (wave-local)
        for (int i = lane * 4; i < 16 * DD; i += 128) {
            int r = i >> 7, c = i & 127;
            *(float4*)&stg[wave][r][c] = *(const float4*)&rpe[(rb + r) * DD + c];
        }
        asm volatile("s_wait_dscnt 0x0" ::: "memory");   // wave-local LDS RAW

        v8f acc = {};
        for (int ks = 0; ks < 4; ++ks) {
            v16h af;
            for (int j = 0; j < 16; ++j)
                af[j] = (_Float16)stg[wave][lm][ks * 32 + frag_k(j, lh)];
            acc = wmma_f16(af, bf[ks], acc);
        }
        if (lm < 8) {
            int layer = lm >> 2, h = lm & 3;
            float bb = (layer == 0) ? b0[h] : b1[h];
            for (int r = 0; r < 8; ++r) {
                int mm = m0 + r + lh * 8;
                bias_all[(long)layer * BIAS1 +
                         (((long)(b * HH + h) * LL + l) * LL) + mm] = acc[r] + bb;
            }
        }
    }
}

// =====================================================================
// 2) fused LayerNorm + QKV projection (M=1024, N=384 fused, K=128)
// =====================================================================
__global__ __launch_bounds__(128) void ln_qkv_kernel(
    const float* __restrict__ x, const float* __restrict__ g, const float* __restrict__ beta,
    const _Float16* __restrict__ Ws,
    const float* __restrict__ bq, const float* __restrict__ bk, const float* __restrict__ bv,
    float* __restrict__ qo, float* __restrict__ ko, float* __restrict__ vo)
{
    __shared__ _Float16 xt[16][DD];
    __shared__ float part[16][8][2];
    __shared__ float stat[16][2];

    int tid = threadIdx.x, wave = tid >> 5, lane = tid & 31;
    int lm = lane & 15, lh = lane >> 4;
    int rowbase = blockIdx.x * 16;
    int row = tid >> 3, seg = tid & 7;

    float vals[16];
    {
        const float* xr = x + (long)(rowbase + row) * DD + seg * 16;
        float s = 0.f, ss = 0.f;
        for (int c = 0; c < 16; ++c) { float t = xr[c]; vals[c] = t; s += t; ss += t * t; }
        part[row][seg][0] = s; part[row][seg][1] = ss;
    }
    __syncthreads();
    if (tid < 16) {
        float s2 = 0.f, ss2 = 0.f;
        for (int k2 = 0; k2 < 8; ++k2) { s2 += part[tid][k2][0]; ss2 += part[tid][k2][1]; }
        float m = s2 / 128.f;
        stat[tid][0] = m;
        stat[tid][1] = rsqrtf(ss2 / 128.f - m * m + 1e-5f);
    }
    __syncthreads();
    {
        float m = stat[row][0], rs = stat[row][1];
        for (int c = 0; c < 16; ++c) {
            int col = seg * 16 + c;
            xt[row][col] = (_Float16)((vals[c] - m) * rs * g[col] + beta[col]);
        }
    }
    __syncthreads();

    // k-outer / n-inner: A fragment gathered once, 6 B tiles per wave
    v8f acc[6] = {};
    for (int ks = 0; ks < 4; ++ks) {
        v16h af;
        for (int j = 0; j < 16; ++j) af[j] = xt[lm][ks * 32 + frag_k(j, lh)];
        for (int t = 0; t < 6; ++t) {
            int nt = wave * 6 + t;                    // 24 tiles over N=384
            v16h bfr = *(const v16h*)(Ws + ((long)(ks * 24 + nt) * 512) + lane * 16);
            acc[t] = wmma_f16(af, bfr, acc[t]);
        }
    }
    for (int t = 0; t < 6; ++t) {
        int nt = wave * 6 + t;
        int which = nt >> 3, ncol = (nt & 7) * 16;
        const float* Bv = (which == 0) ? bq : (which == 1) ? bk : bv;
        float*       O  = (which == 0) ? qo : (which == 1) ? ko : vo;
        float bb = Bv[ncol + lm];
        for (int r = 0; r < 8; ++r)
            O[(long)(rowbase + r + lh * 8) * DD + ncol + lm] = acc[t][r] + bb;
    }
}

// =====================================================================
// 3) attention: K/V staged via LDS in 256-row chunks (coalesced b128),
//    scores via WMMA, LDS softmax, P.V k-split with ds_add_f32 reduce.
// =====================================================================
__global__ __launch_bounds__(128) void attn_kernel(
    const float* __restrict__ q, const float* __restrict__ k,
    const float* __restrict__ v, const float* __restrict__ bias,
    float* __restrict__ out)
{
    __shared__ float    sc[16][LL];      // 32 KB scores / probabilities
    __shared__ _Float16 kv[256][HD];     // 16 KB K or V chunk
    __shared__ _Float16 qt[16][HD];
    __shared__ float red[16][8];
    __shared__ float rowmax[16], rowsum[16];
    __shared__ float oacc[16][HD];

    int wg = blockIdx.x;                 // 256 = B*H*(L/16)
    int qb = wg & 31, h = (wg >> 5) & 3, b = wg >> 7;
    int q0 = qb * 16;
    int tid = threadIdx.x, wave = tid >> 5, lane = tid & 31;
    int lm = lane & 15, lh = lane >> 4;
    const float scale = 0.17677669529663687f;   // 1/sqrt(32)

    {   // Q tile, pre-scaled, one float4 per thread
        int i = tid * 4, r = i >> 5, c = i & 31;
        float4 f = *(const float4*)&q[((long)(b * LL + q0 + r)) * DD + h * HD + c];
        v4h hv;
        hv[0] = (_Float16)(f.x * scale); hv[1] = (_Float16)(f.y * scale);
        hv[2] = (_Float16)(f.z * scale); hv[3] = (_Float16)(f.w * scale);
        *(v4h*)&qt[r][c] = hv;
    }
    __syncthreads();
    v16h af0;
    for (int j = 0; j < 16; ++j) af0[j] = qt[lm][frag_k(j, lh)];

    // --- phase A: S = (Q*scale) K^T + bias ---
    for (int c0 = 0; c0 < LL; c0 += 256) {
        __syncthreads();
        for (int i = tid * 4; i < 256 * HD; i += 512) {
            int r = i >> 5, cc = i & 31;
            float4 f = *(const float4*)&k[((long)(b * LL + c0 + r)) * DD + h * HD + cc];
            *(v4h*)&kv[r][cc] = cvt4(f);
        }
        __syncthreads();
        for (int t = 0; t < 4; ++t) {
            int ml = (wave * 4 + t) * 16;            // local m inside chunk
            v16h bfr;
            for (int j = 0; j < 16; ++j) bfr[j] = kv[ml + lm][frag_k(j, lh)];
            v8f acc = {};
            acc = wmma_f16(af0, bfr, acc);
            for (int r = 0; r < 8; ++r) {
                int rr = r + lh * 8;
                int cc = c0 + ml + lm;
                sc[rr][cc] = acc[r] +
                    bias[(((long)(b * HH + h) * LL + (q0 + rr)) * LL) + cc];
            }
        }
    }
    __syncthreads();

    // --- softmax over 512 cols, 8 threads/row ---
    int row = tid >> 3, seg = tid & 7;
    {
        float mx = -3.0e38f;
        for (int c = seg * 64; c < seg * 64 + 64; ++c) mx = fmaxf(mx, sc[row][c]);
        red[row][seg] = mx;
    }
    __syncthreads();
    if (tid < 16) {
        float mx = red[tid][0];
        for (int s = 1; s < 8; ++s) mx = fmaxf(mx, red[tid][s]);
        rowmax[tid] = mx;
    }
    __syncthreads();
    {
        float mx = rowmax[row], sm = 0.f;
        for (int c = seg * 64; c < seg * 64 + 64; ++c) {
            float e = __expf(sc[row][c] - mx);
            sc[row][c] = e;
            sm += e;
        }
        red[row][seg] = sm;
    }
    __syncthreads();
    if (tid < 16) {
        float s = 0.f;
        for (int g2 = 0; g2 < 8; ++g2) s += red[tid][g2];
        rowsum[tid] = s;
    }
    for (int i = tid; i < 16 * HD; i += 128) oacc[i >> 5][i & 31] = 0.f;

    // --- phase B: O = P V, wave owns 64 k-rows per chunk, acc across chunks ---
    v8f pacc[2] = {};
    for (int c0 = 0; c0 < LL; c0 += 256) {
        __syncthreads();
        for (int i = tid * 4; i < 256 * HD; i += 512) {
            int r = i >> 5, cc = i & 31;
            float4 f = *(const float4*)&v[((long)(b * LL + c0 + r)) * DD + h * HD + cc];
            *(v4h*)&kv[r][cc] = cvt4(f);
        }
        __syncthreads();
        for (int ks = 0; ks < 2; ++ks) {
            int kbl = wave * 64 + ks * 32;           // local k inside chunk
            v16h af;
            for (int j = 0; j < 16; ++j)
                af[j] = (_Float16)sc[lm][c0 + kbl + frag_k(j, lh)];
            for (int nt = 0; nt < 2; ++nt) {
                v16h bfr;
                for (int j = 0; j < 16; ++j)
                    bfr[j] = kv[kbl + frag_k(j, lh)][nt * 16 + lm];
                pacc[nt] = wmma_f16(af, bfr, pacc[nt]);
            }
        }
    }
    for (int nt = 0; nt < 2; ++nt)
        for (int r = 0; r < 8; ++r)
            atomicAdd(&oacc[r + lh * 8][nt * 16 + lm], pacc[nt][r]);  // ds_add_f32
    __syncthreads();
    for (int i = tid; i < 16 * HD; i += 128) {
        int rr = i >> 5, cc = i & 31;
        out[((long)(b * LL + q0 + rr)) * DD + h * HD + cc] = oacc[rr][cc] / rowsum[rr];
    }
}

// =====================================================================
// 4) generic WMMA GEMM: out = [gelu](A @ W + bias) [+ res], swizzled f16 W
// =====================================================================
template <int K, int N, bool GELU, bool RES>
__global__ __launch_bounds__(128) void gemm_kernel(
    const float* __restrict__ A, const _Float16* __restrict__ Ws,
    const float* __restrict__ bias, const float* __restrict__ res,
    float* __restrict__ out)
{
    __shared__ _Float16 at[16][K];
    constexpr int NT = N / 16, WT = NT / 4;
    int tid = threadIdx.x, wave = tid >> 5, lane = tid & 31;
    int lm = lane & 15, lh = lane >> 4;
    long rowbase = (long)blockIdx.x * 16;

    for (int i = tid * 4; i < 16 * K; i += 512) {
        int r = i / K, c = i % K;
        float4 f = *(const float4*)&A[(rowbase + r) * K + c];
        *(v4h*)&at[r][c] = cvt4(f);
    }
    __syncthreads();

    v8f acc[WT] = {};
    for (int ks = 0; ks < K / 32; ++ks) {
        v16h af;
        for (int j = 0; j < 16; ++j) af[j] = at[lm][ks * 32 + frag_k(j, lh)];
        for (int t = 0; t < WT; ++t) {
            int nt = wave * WT + t;
            v16h bfr = *(const v16h*)(Ws + ((long)(ks * NT + nt) * 512) + lane * 16);
            acc[t] = wmma_f16(af, bfr, acc[t]);
        }
    }
    for (int t = 0; t < WT; ++t) {
        int ncol = (wave * WT + t) * 16;
        float bb = bias[ncol + lm];
        for (int r = 0; r < 8; ++r) {
            long orow = rowbase + r + lh * 8;
            float val = acc[t][r] + bb;
            if constexpr (GELU) val = 0.5f * val * (1.f + erff(val * 0.70710678118f));
            if constexpr (RES)  val += res[orow * N + ncol + lm];
            out[orow * N + ncol + lm] = val;
        }
    }
}

// =====================================================================
// 5) standalone LayerNorm over 128 cols (ln2)
// =====================================================================
__global__ __launch_bounds__(128) void ln_kernel(
    const float* __restrict__ x, const float* __restrict__ g,
    const float* __restrict__ beta, float* __restrict__ y)
{
    __shared__ float part[16][8][2];
    __shared__ float stat[16][2];
    int tid = threadIdx.x;
    int rowbase = blockIdx.x * 16;
    int row = tid >> 3, seg = tid & 7;
    const float* xr = x + (long)(rowbase + row) * DD + seg * 16;
    float vals[16];
    float s = 0.f, ss = 0.f;
    for (int c = 0; c < 16; ++c) { float t = xr[c]; vals[c] = t; s += t; ss += t * t; }
    part[row][seg][0] = s; part[row][seg][1] = ss;
    __syncthreads();
    if (tid < 16) {
        float s2 = 0.f, ss2 = 0.f;
        for (int k2 = 0; k2 < 8; ++k2) { s2 += part[tid][k2][0]; ss2 += part[tid][k2][1]; }
        float m = s2 / 128.f;
        stat[tid][0] = m;
        stat[tid][1] = rsqrtf(ss2 / 128.f - m * m + 1e-5f);
    }
    __syncthreads();
    float m = stat[row][0], rs = stat[row][1];
    for (int c = 0; c < 16; ++c) {
        int col = seg * 16 + c;
        y[(long)(rowbase + row) * DD + col] = (vals[c] - m) * rs * g[col] + beta[col];
    }
}

// =====================================================================
// 6) conv1d (k=3,s=2,p=1) + LN: x[2,512,128] -> [2,256,256]
//    uniform per-run bounds, float4 loads, K=384, N=256
// =====================================================================
__global__ __launch_bounds__(128) void conv1d_kernel(
    const float* __restrict__ x, const _Float16* __restrict__ Ws,
    const float* __restrict__ g, const float* __restrict__ beta,
    float* __restrict__ out)
{
    __shared__ _Float16 at[16][384];
    __shared__ float yt[16][256];
    __shared__ float part[16][8][2];
    __shared__ float stat[16][2];
    int tid = threadIdx.x, wave = tid >> 5, lane = tid & 31;
    int lm = lane & 15, lh = lane >> 4;
    int rowbase = blockIdx.x * 16;     // rows: b*256+lo

    // 48 runs of 128 channels; bounds uniform per run; one float4 per lane
    for (int run = wave; run < 48; run += 4) {
        int r = run / 3, kw = run % 3;
        int grow = rowbase + r, b = grow >> 8, lo = grow & 255;
        int pos = 2 * lo + kw - 1;
        v4h hv = {};
        if (pos >= 0 && pos < LL)
            hv = cvt4(*(const float4*)&x[((long)(b * LL + pos)) * DD + lane * 4]);
        *(v4h*)&at[r][kw * 128 + lane * 4] = hv;
    }
    __syncthreads();

    v8f acc[4] = {};
    for (int ks = 0; ks < 12; ++ks) {
        v16h af;
        for (int j = 0; j < 16; ++j) af[j] = at[lm][ks * 32 + frag_k(j, lh)];
        for (int t = 0; t < 4; ++t) {
            int nt = wave * 4 + t;
            v16h bfr = *(const v16h*)(Ws + ((long)(ks * 16 + nt) * 512) + lane * 16);
            acc[t] = wmma_f16(af, bfr, acc[t]);
        }
    }
    for (int t = 0; t < 4; ++t)
        for (int r = 0; r < 8; ++r)
            yt[r + lh * 8][(wave * 4 + t) * 16 + lm] = acc[t][r];
    __syncthreads();

    int row = tid >> 3, seg = tid & 7;
    float s = 0.f, ss = 0.f;
    for (int c = seg * 32; c < seg * 32 + 32; ++c) { float t = yt[row][c]; s += t; ss += t * t; }
    part[row][seg][0] = s; part[row][seg][1] = ss;
    __syncthreads();
    if (tid < 16) {
        float s2 = 0.f, ss2 = 0.f;
        for (int k2 = 0; k2 < 8; ++k2) { s2 += part[tid][k2][0]; ss2 += part[tid][k2][1]; }
        float m = s2 / 256.f;
        stat[tid][0] = m;
        stat[tid][1] = rsqrtf(ss2 / 256.f - m * m + 1e-5f);
    }
    __syncthreads();
    float m = stat[row][0], rs = stat[row][1];
    for (int c = seg * 32; c < seg * 32 + 32; ++c)
        out[((long)(rowbase + row)) * 256 + c] = (yt[row][c] - m) * rs * g[c] + beta[c];
}

// =====================================================================
// 7) conv2d (3x3,s=2,p=1) + LN on rpe: the 268MB implicit-GEMM pass
//    M = 2*256*256, N = 256, K = 1152; uniform per-run bounds + b128 loads
// =====================================================================
__global__ __launch_bounds__(128) void conv2d_kernel(
    const float* __restrict__ rpe, const _Float16* __restrict__ Ws,
    const float* __restrict__ g, const float* __restrict__ beta,
    float* __restrict__ out)
{
    __shared__ _Float16 at[16][1152];   // 36 KB patch tile
    __shared__ float yt[16][256];       // 16 KB output tile
    __shared__ float part[16][8][2];
    __shared__ float stat[16][2];
    int tid = threadIdx.x, wave = tid >> 5, lane = tid & 31;
    int lm = lane & 15, lh = lane >> 4;
    long rowbase = (long)blockIdx.x * 16;   // row = (b*256+p)*256 + q
    long bp = rowbase >> 8;
    int b = (int)(bp >> 8), p = (int)(bp & 255), q0 = (int)(rowbase & 255);

    // 144 runs (16 rows x 9 taps) of 128 channels; bounds uniform per run
    for (int run = wave; run < 144; run += 4) {
        int r = run / 9, tap = run % 9, kh = tap / 3, kw = tap % 3;
        int row1 = 2 * p + kw - 1;
        int row2 = 2 * (q0 + r) + kh - 1;
        v4h hv = {};
        if (row1 >= 0 && row1 < LL && row2 >= 0 && row2 < LL) {
            const float* src = &rpe[(((long)(b * LL + row1)) * LL + row2) * DD + lane * 4];
            __builtin_prefetch(src + 2 * LL * DD, 0, 1);   // global_prefetch_b8
            hv = cvt4(*(const float4*)src);
        }
        *(v4h*)&at[r][tap * 128 + lane * 4] = hv;
    }
    __syncthreads();

    v8f acc[4] = {};
    for (int ks = 0; ks < 36; ++ks) {
        v16h af;
        for (int j = 0; j < 16; ++j) af[j] = at[lm][ks * 32 + frag_k(j, lh)];
        for (int t = 0; t < 4; ++t) {
            int nt = wave * 4 + t;
            v16h bfr = *(const v16h*)(Ws + ((long)(ks * 16 + nt) * 512) + lane * 16);
            acc[t] = wmma_f16(af, bfr, acc[t]);
        }
    }
    for (int t = 0; t < 4; ++t)
        for (int r = 0; r < 8; ++r)
            yt[r + lh * 8][(wave * 4 + t) * 16 + lm] = acc[t][r];
    __syncthreads();

    int row = tid >> 3, seg = tid & 7;
    float s = 0.f, ss = 0.f;
    for (int c = seg * 32; c < seg * 32 + 32; ++c) { float t = yt[row][c]; s += t; ss += t * t; }
    part[row][seg][0] = s; part[row][seg][1] = ss;
    __syncthreads();
    if (tid < 16) {
        float s2 = 0.f, ss2 = 0.f;
        for (int k2 = 0; k2 < 8; ++k2) { s2 += part[tid][k2][0]; ss2 += part[tid][k2][1]; }
        float m = s2 / 256.f;
        stat[tid][0] = m;
        stat[tid][1] = rsqrtf(ss2 / 256.f - m * m + 1e-5f);
    }
    __syncthreads();
    float m = stat[row][0], rs = stat[row][1];
    for (int c = seg * 32; c < seg * 32 + 32; ++c)
        out[(rowbase + row) * 256 + c] = (yt[row][c] - m) * rs * g[c] + beta[c];
}

// =====================================================================
// host launcher
// =====================================================================
extern "C" void kernel_launch(void* const* d_in, const int* in_sizes, int n_in,
                              void* d_out, int out_size, void* d_ws, size_t ws_size,
                              hipStream_t stream) {
    const float* x_in = (const float*)d_in[0];
    const float* rpe  = (const float*)d_in[1];
    // params: per-layer base 2 + l*18
    // 0 ln1_g 1 ln1_b 2 wq 3 bq 4 wk 5 bk 6 wv 7 bv 8 w_rpe 9 b_rpe
    // 10 wo 11 bo 12 ln2_g 13 ln2_b 14 w1 15 b1 16 w2 17 b2
    const float* ds_w  = (const float*)d_in[38];
    const float* ds_g  = (const float*)d_in[39];
    const float* ds_b  = (const float*)d_in[40];
    const float* rds_w = (const float*)d_in[41];
    const float* rds_g = (const float*)d_in[42];
    const float* rds_b = (const float*)d_in[43];

    float* ws = (float*)d_ws;
    float* q        = ws;
    float* k        = q + NX;
    float* v        = k + NX;
    float* attn     = v + NX;
    float* hbuf     = attn + NX;
    float* h2       = hbuf + NX;
    float* hid      = h2 + NX;                 // B*L*HID = 524288
    float* xa       = hid + (long)BB * LL * HID;
    float* bias_all = xa + NX;                 // 2 * BIAS1 floats

    // swizzled f16 weight region (32B aligned: float region is 32B multiple)
    _Float16* hws = (_Float16*)(bias_all + 2 * BIAS1);
    _Float16* qkv_s[2] = { hws + 0,      hws + 49152 };
    _Float16* wo_s[2]  = { hws + 98304,  hws + 114688 };
    _Float16* w1_s[2]  = { hws + 131072, hws + 196608 };
    _Float16* w2_s[2]  = { hws + 262144, hws + 327680 };
    _Float16* rpe_s    = hws + 393216;
    _Float16* c1_s     = hws + 395264;
    _Float16* c2_s     = hws + 493568;

    float* out      = (float*)d_out;
    float* conv_out = out;                      // x_  [2,256,256]
    float* x_out    = out + 2L * 256 * 256;     // x   [2,512,128]
    float* rpe_out  = x_out + NX;               // rpe_[2,256,256,256]

    // ---- weight pre-swizzle (f16 fragment-major) ----
    for (int l = 0; l < 2; ++l) {
        const int base = 2 + l * 18;
        auto P = [&](int i) { return (const float*)d_in[base + i]; };
        prep_w_kernel<<<192, 256, 0, stream>>>(1, P(2), P(4), P(6), qkv_s[l], 128, 384);
        prep_w_kernel<<<64,  256, 0, stream>>>(0, P(10), nullptr, nullptr, wo_s[l], 128, 128);
        prep_w_kernel<<<256, 256, 0, stream>>>(0, P(14), nullptr, nullptr, w1_s[l], 128, 512);
        prep_w_kernel<<<256, 256, 0, stream>>>(0, P(16), nullptr, nullptr, w2_s[l], 512, 128);
    }
    prep_w_kernel<<<8,    256, 0, stream>>>(2, (const float*)d_in[2 + 8],
                                            (const float*)d_in[2 + 18 + 8], nullptr,
                                            rpe_s, 128, 16);
    prep_w_kernel<<<384,  256, 0, stream>>>(3, ds_w, nullptr, nullptr, c1_s, 384, 256);
    prep_w_kernel<<<1152, 256, 0, stream>>>(4, rds_w, nullptr, nullptr, c2_s, 1152, 256);

    // ---- one pass over rpe -> bias for BOTH layers ----
    bias_proj_kernel<<<2048, 128, 0, stream>>>(
        rpe, rpe_s, (const float*)d_in[2 + 9], (const float*)d_in[2 + 18 + 9],
        bias_all);

    const float* cur = x_in;
    for (int l = 0; l < 2; ++l) {
        const int base = 2 + l * 18;
        auto P = [&](int i) { return (const float*)d_in[base + i]; };
        ln_qkv_kernel<<<64, 128, 0, stream>>>(cur, P(0), P(1), qkv_s[l],
                                              P(3), P(5), P(7), q, k, v);
        attn_kernel<<<256, 128, 0, stream>>>(q, k, v, bias_all + (long)l * BIAS1,
                                             attn);
        gemm_kernel<128, 128, false, true><<<64, 128, 0, stream>>>(
            attn, wo_s[l], P(11), cur, hbuf);                     // wo + shortcut
        ln_kernel<<<64, 128, 0, stream>>>(hbuf, P(12), P(13), h2);
        gemm_kernel<128, 512, true, false><<<64, 128, 0, stream>>>(
            h2, w1_s[l], P(15), nullptr, hid);                    // fc1 + gelu
        float* xnext = (l == 0) ? xa : x_out;   // last layer writes the x output
        gemm_kernel<512, 128, false, true><<<64, 128, 0, stream>>>(
            hid, w2_s[l], P(17), cur, xnext);                     // fc2 + shortcut
        cur = xnext;
    }

    conv1d_kernel<<<32, 128, 0, stream>>>(x_out, c1_s, ds_g, ds_b, conv_out);
    conv2d_kernel<<<8192, 128, 0, stream>>>(rpe, c2_s, rds_g, rds_b, rpe_out);
}
